// LSTMEncoder_32530082300436
// MI455X (gfx1250) — matrix operand order repaired and checked by reading, not compiled, not run on hardware
//
#include <hip/hip_runtime.h>

// Fused 2-layer LSTM, last-timestep output. gfx1250 wave32 + WMMA f16 16x16x32.
// B=4096, T=512, D_IN=6, H=32, 4H=128. One wave owns 16 batch rows and carries
// (h0,c0,h1,c1) through all 512 steps; h_seq is never materialized.

typedef __attribute__((ext_vector_type(16))) _Float16 v16h;
typedef __attribute__((ext_vector_type(8)))  float    v8f;

#define B_TOT 4096
#define T_LEN 512
#define D_IN  6
#define H_DIM 32
#define G4H   128   // 4*H
#define NTILE 8     // G4H / 16
#define WAVES 2     // waves per block

struct alignas(32) Frag16 { uint4 lo, hi; };

static __device__ __forceinline__ v16h ld_frag_pair(const _Float16* p0, const _Float16* p1) {
  Frag16 f;
  f.lo = *reinterpret_cast<const uint4*>(p0);
  f.hi = *reinterpret_cast<const uint4*>(p1);
  return __builtin_bit_cast(v16h, f);
}

// A fragment (16x32 f16) from an LDS tile stored row-major [16][32].
// Contiguous k order used here (matches B staging): lanes 0-15 hold M=lane, K = 0..15; lanes 16-31 hold
// M=lane-16, K = 16..31.
static __device__ __forceinline__ v16h ld_frag_a(const _Float16* tile, int lane) {
  const int m  = lane & 15;
  const int hi = (lane >> 4) & 1;
  const _Float16* p = tile + m * 32 + hi * 16;
  return ld_frag_pair(p, p + 8);
}

// B fragment (32x16 f16): pre-swizzled in LDS so each lane's 16 halves are
// contiguous (lane L: N=L%16, K = L<16 ? 0..15 : 16..31).
static __device__ __forceinline__ v16h ld_frag_b(const _Float16* base, int n, int lane) {
  const _Float16* p = base + (n * 32 + lane) * 16;
  return ld_frag_pair(p, p + 8);
}

// gfx1250 has a native TRANS-unit tanh (v_tanh_f32). Use it when the builtin
// is declared; otherwise fall back to the exp-based formulation.
static __device__ __forceinline__ float ftanh(float x) {
#if defined(__has_builtin) && __has_builtin(__builtin_amdgcn_tanhf)
  return __builtin_amdgcn_tanhf(x);
#else
  return 2.0f / (1.0f + __expf(-2.0f * x)) - 1.0f;
#endif
}
// sigmoid(x) = 0.5*tanh(x/2) + 0.5  -> 1 trans op + mul + fma
static __device__ __forceinline__ float fsig(float x) {
  return __builtin_fmaf(0.5f, ftanh(0.5f * x), 0.5f);
}

static __device__ __forceinline__ v8f splat8(float v) {
  v8f r;
  #pragma unroll
  for (int i = 0; i < 8; ++i) r[i] = v;
  return r;
}

// Elementwise LSTM cell update on one half (16 cols) of the hidden state.
// Gate accumulators share the exact WMMA D-layout, so this is register-only.
static __device__ __forceinline__ void lstm_elem(const v8f& gi, const v8f& gf,
                                                 const v8f& gg, const v8f& go,
                                                 v8f& c, v8f& h) {
  #pragma unroll
  for (int r = 0; r < 8; ++r) {
    float i = fsig(gi[r]);
    float f = fsig(gf[r]);
    float g = ftanh(gg[r]);
    float o = fsig(go[r]);
    float cn = __builtin_fmaf(f, c[r], i * g);
    c[r] = cn;
    h[r] = o * ftanh(cn);
  }
}

__global__ __launch_bounds__(WAVES * 32)
void lstm2_fused_wmma(const float* __restrict__ x,
                      const float* __restrict__ Wih0, const float* __restrict__ Whh0,
                      const float* __restrict__ bih0, const float* __restrict__ bhh0,
                      const float* __restrict__ Wih1, const float* __restrict__ Whh1,
                      const float* __restrict__ bih1, const float* __restrict__ bhh1,
                      float* __restrict__ out) {
  // 4 weight matrices in B-fragment order, f16, K padded to 32: 4 x 8 KB.
  __shared__ __align__(32) _Float16 ldsW[4][NTILE * 32 * 16];
  // Per-wave h0/h1 staging tiles [16][32] f16 for D-layout -> A-layout bounce.
  __shared__ __align__(32) _Float16 ldsH[WAVES][2][16 * 32];

  const int tid  = threadIdx.x;
  const int lane = tid & 31;
  const int wave = tid >> 5;
  const int b0   = (blockIdx.x * WAVES + wave) * 16;

  // ---- stage weights (global f32 -> LDS f16, fragment order, zero-pad K) ----
  {
    const float* src[4] = {Wih0, Whh0, Wih1, Whh1};
    const int    ks [4] = {D_IN, H_DIM, H_DIM, H_DIM};
    #pragma unroll
    for (int m = 0; m < 4; ++m) {
      for (int e = tid; e < G4H * 32; e += (int)blockDim.x) {
        int k = e & 31;
        int g = e >> 5;                       // gate row 0..127
        float v = (k < ks[m]) ? src[m][g * ks[m] + k] : 0.0f;
        int n  = g >> 4;                      // N tile
        int bl = (g & 15) + (k & 16);         // fragment lane
        ldsW[m][(n * 32 + bl) * 16 + (k & 15)] = (_Float16)v;
      }
    }
  }
  __syncthreads();

  // Per-lane gate biases (column = n*16 + lane%16; identical for both lane halves).
  float bias0[NTILE], bias1[NTILE];
  #pragma unroll
  for (int n = 0; n < NTILE; ++n) {
    int col = n * 16 + (lane & 15);
    bias0[n] = bih0[col] + bhh0[col];
    bias1[n] = bih1[col] + bhh1[col];
  }

  v16h h0f = {};                 // A-fragment of layer-0 hidden state
  v16h h1f = {};                 // A-fragment of layer-1 hidden state
  v8f c0l = {}, c0h = {}, c1l = {}, c1h = {};
  v8f h1l = {}, h1h = {};        // layer-1 h in D-layout (final output)

  _Float16* hs0 = &ldsH[wave][0][0];
  _Float16* hs1 = &ldsH[wave][1][0];
  const int jm = lane & 15;            // local column
  const int mo = (lane >> 4) << 3;     // +8 rows for upper lane half

  for (int t = 0; t < T_LEN; ++t) {
    // x_t A-fragment: K padded to 32; only K<6 nonzero -> lanes 0-15 carry data.
    v16h xf = {};
    if (lane < 16) {
      const float* px = x + ((size_t)(b0 + lane) * T_LEN + t) * D_IN;
      float2 x01 = *reinterpret_cast<const float2*>(px + 0);
      float2 x23 = *reinterpret_cast<const float2*>(px + 2);
      float2 x45 = *reinterpret_cast<const float2*>(px + 4);
      xf[0] = (_Float16)x01.x; xf[1] = (_Float16)x01.y;
      xf[2] = (_Float16)x23.x; xf[3] = (_Float16)x23.y;
      xf[4] = (_Float16)x45.x; xf[5] = (_Float16)x45.y;
      if (t + 1 < T_LEN) __builtin_prefetch(px + D_IN, 0, 1);  // global_prefetch_b8
    }

    // ----- layer 0: gates = x_t@Wih0^T + h0@Whh0^T + b -----
    v8f acc0[NTILE];
    #pragma unroll
    for (int n = 0; n < NTILE; ++n) {
      v8f a = splat8(bias0[n]);
      a = __builtin_amdgcn_wmma_f32_16x16x32_f16(false, xf,  false,
              ld_frag_b(ldsW[0], n, lane), (short)0, a, false, false);
      a = __builtin_amdgcn_wmma_f32_16x16x32_f16(false, h0f, false,
              ld_frag_b(ldsW[1], n, lane), (short)0, a, false, false);
      acc0[n] = a;
    }
    v8f h0l, h0h;
    lstm_elem(acc0[0], acc0[2], acc0[4], acc0[6], c0l, h0l);   // cols 0..15
    lstm_elem(acc0[1], acc0[3], acc0[5], acc0[7], c0h, h0h);   // cols 16..31

    // Stage h0 (f16 [16][32]) and reload in A-fragment order (wave-private LDS).
    #pragma unroll
    for (int r = 0; r < 8; ++r) {
      hs0[(r + mo) * 32 + jm]      = (_Float16)h0l[r];
      hs0[(r + mo) * 32 + jm + 16] = (_Float16)h0h[r];
    }
    h0f = ld_frag_a(hs0, lane);

    // ----- layer 1: gates = h0@Wih1^T + h1@Whh1^T + b -----
    v8f acc1[NTILE];
    #pragma unroll
    for (int n = 0; n < NTILE; ++n) {
      v8f a = splat8(bias1[n]);
      a = __builtin_amdgcn_wmma_f32_16x16x32_f16(false, h0f, false,
              ld_frag_b(ldsW[2], n, lane), (short)0, a, false, false);
      a = __builtin_amdgcn_wmma_f32_16x16x32_f16(false, h1f, false,
              ld_frag_b(ldsW[3], n, lane), (short)0, a, false, false);
      acc1[n] = a;
    }
    lstm_elem(acc1[0], acc1[2], acc1[4], acc1[6], c1l, h1l);
    lstm_elem(acc1[1], acc1[3], acc1[5], acc1[7], c1h, h1h);

    #pragma unroll
    for (int r = 0; r < 8; ++r) {
      hs1[(r + mo) * 32 + jm]      = (_Float16)h1l[r];
      hs1[(r + mo) * 32 + jm + 16] = (_Float16)h1h[r];
    }
    h1f = ld_frag_a(hs1, lane);
  }

  // Final h1 (t = T-1) -> out[B][H], written straight from the D-layout regs.
  #pragma unroll
  for (int r = 0; r < 8; ++r) {
    int row = b0 + r + mo;
    out[row * H_DIM + jm]      = h1l[r];
    out[row * H_DIM + jm + 16] = h1h[r];
  }
}

extern "C" void kernel_launch(void* const* d_in, const int* in_sizes, int n_in,
                              void* d_out, int out_size, void* d_ws, size_t ws_size,
                              hipStream_t stream) {
  (void)in_sizes; (void)n_in; (void)out_size; (void)d_ws; (void)ws_size;
  const float* x    = (const float*)d_in[0];
  const float* Wih0 = (const float*)d_in[1];
  const float* Whh0 = (const float*)d_in[2];
  const float* bih0 = (const float*)d_in[3];
  const float* bhh0 = (const float*)d_in[4];
  const float* Wih1 = (const float*)d_in[5];
  const float* Whh1 = (const float*)d_in[6];
  const float* bih1 = (const float*)d_in[7];
  const float* bhh1 = (const float*)d_in[8];
  float* out = (float*)d_out;

  dim3 grid(B_TOT / (16 * WAVES));   // 128 workgroups, 256 wave-tiles total
  dim3 block(WAVES * 32);
  hipLaunchKernelGGL(lstm2_fused_wmma, grid, block, 0, stream,
                     x, Wih0, Whh0, bih0, bhh0, Wih1, Whh1, bih1, bhh1, out);
}
